// Correlation_72430328481456
// MI455X (gfx1250) — matrix operand order repaired and checked by reading, not compiled, not run on hardware
//
#include <hip/hip_runtime.h>

// FlowNet correlation on MI455X (gfx1250, wave32, WMMA).
// out[b, dyi*21+dxi, y, x] = (1/256) * sum_c in1[b,c,y,x] * in2pad[b,c,y+dy,x+dx]
// dy = 2*dyi-20, dx = 2*dxi-20.  B=8 C=256 H=96 W=128, D=441.

typedef __attribute__((ext_vector_type(16))) _Float16 v16h;
typedef __attribute__((ext_vector_type(8)))  float    v8f;

#define NB   8
#define NC   256
#define NH   96
#define NW   128
#define NG   21        // displacements per axis
#define ND   441
#define CP   260       // padded channel stride in LDS (f16 elems); 520B -> 8B-aligned runs
#define XPN  192       // staged in2 columns: x' = xp-20, covers all N-tile columns
#define CSP  132       // padded Cs row stride (f32) -> conflict-free scatter

union H16 { v16h h; uint32_t u[8]; };

__device__ __forceinline__ void ld8(H16& f, const uint32_t* p) {
  f.u[0] = p[0]; f.u[1] = p[1]; f.u[2] = p[2]; f.u[3] = p[3];
  f.u[4] = p[4]; f.u[5] = p[5]; f.u[6] = p[6]; f.u[7] = p[7];
}

__global__ __launch_bounds__(256) void corr_wmma_kernel(
    const float* __restrict__ in1, const float* __restrict__ in2,
    float* __restrict__ out)
{
  const int y    = blockIdx.x;        // 0..95
  const int b    = blockIdx.y;        // 0..7
  const int tid  = threadIdx.x;       // 256 threads = 8 waves
  const int lane = tid & 31;
  const int wave = tid >> 5;

  extern __shared__ char smem[];
  _Float16* As = (_Float16*)smem;             // [128][CP] f16 : in1 row, [x][c]
  _Float16* Bs = As + NW * CP;                // [XPN][CP] f16 : in2 row (shifted), [xp][c]
  float*    Cs = (float*)(Bs + XPN * CP);     // [21][CSP] f32 : per-dy results [dxi][x]

  // ---- stage A: in1[b,:,y,:] -> As[x][c] (coalesced global, f32->f16) ----
  {
    const float* src = in1 + ((size_t)b * NC * NH + (size_t)y) * NW;
    for (int idx = tid; idx < NC * NW; idx += 256) {
      int c = idx >> 7;            // /128
      int x = idx & (NW - 1);
      As[x * CP + c] = (_Float16)src[(size_t)c * (NH * NW) + x];
    }
  }
  __syncthreads();

  // ---- per-wave M-tile: 16 same-parity x values; load A fragments ONCE ----
  // wave w: x0 = (w>>1)*32, parity p = w&1, rows x = x0 + 2m + p, m=0..15
  const int x0  = (wave >> 1) * 32;
  const int par = wave & 1;
  const int m   = lane & 15;
  const int hi  = lane >> 4;          // lane half
  H16 afr[8];                         // 8 K-blocks of 32 channels
  {
    const int xA = x0 + 2 * m + par;
    const int kh = hi << 4;           // 0 or 16 (A layout half-select)
    const uint32_t* arow = (const uint32_t*)(As + (size_t)xA * CP);
    #pragma unroll
    for (int k8 = 0; k8 < 8; ++k8) {
      int c0 = (k8 << 5) + kh;                      // kb + kh (even)
      const uint32_t* p0 = arow + (c0 >> 1);        // K = c0 .. c0+15
      const uint32_t* p1 = p0 + 4;
      afr[k8].u[0] = p0[0]; afr[k8].u[1] = p0[1];
      afr[k8].u[2] = p0[2]; afr[k8].u[3] = p0[3];
      afr[k8].u[4] = p1[0]; afr[k8].u[5] = p1[1];
      afr[k8].u[6] = p1[2]; afr[k8].u[7] = p1[3];
    }
  }

  for (int dyi = 0; dyi < NG; ++dyi) {
    const int y2 = y + 2 * dyi - 20;
    float* orow = out + (((size_t)b * ND + (size_t)dyi * NG) * NH + y) * NW;

    if (y2 < 0 || y2 >= NH) {         // uniform over block: whole dy slab is zero
      for (int idx = tid; idx < NG * NW; idx += 256) {
        int j = idx >> 7, x = idx & (NW - 1);
        orow[(size_t)j * (NH * NW) + x] = 0.0f;
      }
      continue;
    }

    // ---- stage B: in2[b,:,y2,xp-20] -> Bs[xp][c], zero-fill borders ----
    {
      const float* src = in2 + ((size_t)b * NC * NH + (size_t)y2) * NW;
      for (int idx = tid; idx < NC * XPN; idx += 256) {
        int c  = idx / XPN;
        int xp = idx - c * XPN;
        int xg = xp - 20;
        float v = (xg >= 0 && xg < NW) ? src[(size_t)c * (NH * NW) + xg] : 0.0f;
        Bs[xp * CP + c] = (_Float16)v;
      }
    }
    __syncthreads();

    // ---- band GEMM: 3 N-tiles of 16 same-parity x' columns, K=256 ----
    #pragma unroll
    for (int nt = 0; nt < 3; ++nt) {
      v8f acc = {0.f, 0.f, 0.f, 0.f, 0.f, 0.f, 0.f, 0.f};
      const int ncol = nt * 16 + (lane & 15);     // column index (step-2 x')
      const int xpB  = x0 + par + 2 * ncol;       // = x' + 20  (fits in [0,XPN))
      const int kh2  = hi << 4;                   // 0 or 16 (B layout half-select)
      const uint32_t* brow =
          (const uint32_t*)(Bs + (size_t)xpB * CP) + (kh2 >> 1);

      // software-pipelined B-fragment loads: double buffer so the DS load for
      // step k+1 is in flight while the WMMA for step k executes
      H16 bf[2];
      ld8(bf[0], brow);                            // K-block 0
      #pragma unroll
      for (int k8 = 0; k8 < 8; ++k8) {
        if (k8 < 7) ld8(bf[(k8 + 1) & 1], brow + ((k8 + 1) << 4));
        acc = __builtin_amdgcn_wmma_f32_16x16x32_f16(
            false, afr[k8].h, false, bf[k8 & 1].h, (short)0, acc, false, false);
      }

      // scatter the band: dx index j = ncol - m in [0,20]
      #pragma unroll
      for (int v = 0; v < 8; ++v) {
        int mm = v + (hi << 3);                   // C/D layout: M = v + 8*hi
        int j  = ncol - mm;
        if (j >= 0 && j < NG)
          Cs[j * CSP + x0 + 2 * mm + par] = acc[v];
      }
    }
    __syncthreads();

    // ---- coalesced write-out of 21 rows, scaled by 1/C ----
    for (int idx = tid; idx < NG * NW; idx += 256) {
      int j = idx >> 7, x = idx & (NW - 1);
      orow[(size_t)j * (NH * NW) + x] = Cs[j * CSP + x] * (1.0f / 256.0f);
    }
    // next iteration's staging is ordered by its own __syncthreads()
  }
}

extern "C" void kernel_launch(void* const* d_in, const int* in_sizes, int n_in,
                              void* d_out, int out_size, void* d_ws, size_t ws_size,
                              hipStream_t stream) {
  (void)in_sizes; (void)n_in; (void)d_ws; (void)ws_size; (void)out_size;
  const float* in1 = (const float*)d_in[0];
  const float* in2 = (const float*)d_in[1];
  float* out = (float*)d_out;

  const size_t shmem = (size_t)(NW * CP + XPN * CP) * sizeof(_Float16)
                     + (size_t)NG * CSP * sizeof(float);   // 177,488 B < 320 KB
  (void)hipFuncSetAttribute(reinterpret_cast<const void*>(corr_wmma_kernel),
                            hipFuncAttributeMaxDynamicSharedMemorySize,
                            (int)shmem);
  dim3 grid(NH, NB);   // one workgroup per (y, b)
  corr_wmma_kernel<<<grid, 256, shmem, stream>>>(in1, in2, out);
}
